// ReservoirNet_70540542869626
// MI455X (gfx1250) — matrix operand order repaired and checked by reading, not compiled, not run on hardware
//
#include <hip/hip_runtime.h>
#include <hip/hip_bf16.h>
#include <math.h>

// Problem constants (match reference)
#define EMB    1024
#define NTOK   32000
#define LAYERS 2
#define BATCH  4
#define SEQ    1024
#define KMAX   16    // CONN=8 in reference; pad to 16 slots (zeros are harmless)

#define MROWS  64    // A-tile rows per workgroup (4 M-tiles of 16 per wave)

typedef __bf16 bf16_t;
typedef __attribute__((ext_vector_type(16))) __bf16 v16bf;
typedef __attribute__((ext_vector_type(8)))  float  v8f;
typedef __attribute__((ext_vector_type(4)))  unsigned u32x4;
typedef __attribute__((ext_vector_type(8)))  int      i32x8;
typedef __attribute__((ext_vector_type(4)))  int      i32x4;

union FragBF { v16bf v; uint4 q[2]; };

// ---------------------------------------------------------------------------
// Kernel 1: extract the 8-nonzero-per-row sparse structure of the reservoir
// matrices a[2][1024][1024] into (vals, cols)[2][1024][KMAX].
// ---------------------------------------------------------------------------
__global__ void sparsify_kernel(const float* __restrict__ a,
                                float* __restrict__ vals,
                                int*   __restrict__ cols) {
    int row = blockIdx.x * blockDim.x + threadIdx.x;   // 0 .. LAYERS*EMB-1
    if (row >= LAYERS * EMB) return;
    const float* ar = a + (size_t)row * EMB;
    int cnt = 0;
    for (int j = 0; j < EMB; ++j) {
        float v = ar[j];
        if (v != 0.0f && cnt < KMAX) {
            vals[row * KMAX + cnt] = v;
            cols[row * KMAX + cnt] = j;
            ++cnt;
        }
    }
    for (; cnt < KMAX; ++cnt) {
        vals[row * KMAX + cnt] = 0.0f;
        cols[row * KMAX + cnt] = 0;
    }
}

// ---------------------------------------------------------------------------
// Kernel 2: embedding gather, one float4 per thread.
// ---------------------------------------------------------------------------
__global__ void gather_emb_kernel(const int* __restrict__ x,
                                  const float* __restrict__ emb,
                                  float* __restrict__ xe) {
    size_t i = (size_t)blockIdx.x * blockDim.x + threadIdx.x;   // over B*SEQ*(EMB/4)
    const int per_row = EMB / 4;
    size_t bt = i / per_row;
    int    e4 = (int)(i % per_row);
    int    tok = x[bt];
    ((float4*)xe)[i] = ((const float4*)emb)[(size_t)tok * per_row + e4];
}

// ---------------------------------------------------------------------------
// Kernel 3: f32 -> bf16 conversion (w_out), one float4 per thread.
// ---------------------------------------------------------------------------
__global__ void cvt_bf16_kernel(const float* __restrict__ in,
                                bf16_t* __restrict__ out) {
    size_t i = ((size_t)blockIdx.x * blockDim.x + threadIdx.x) * 4;
    float4 f = *(const float4*)(in + i);
    out[i + 0] = (bf16_t)f.x;
    out[i + 1] = (bf16_t)f.y;
    out[i + 2] = (bf16_t)f.z;
    out[i + 3] = (bf16_t)f.w;
}

// ---------------------------------------------------------------------------
// Kernel 4: one reservoir layer. One workgroup per batch (independent).
// Double-buffered LDS state -> ONE barrier per timestep.
//   h_t[e] = tanh( sum_k vals[e][k] * h_{t-1}[cols[e][k]] + x_t[e] )
// ---------------------------------------------------------------------------
__global__ __launch_bounds__(EMB) void reservoir_layer_kernel(
        const float* __restrict__ xin,      // [B][SEQ][EMB]
        const float* __restrict__ vals,     // [EMB][KMAX] (this layer)
        const int*   __restrict__ cols,     // [EMB][KMAX]
        float*       __restrict__ hout_f32, // [B][SEQ][EMB] (if !write_bf)
        bf16_t*      __restrict__ hout_bf,  // [B][SEQ][EMB] (if write_bf)
        int write_bf) {
    __shared__ float hs[2][EMB];
    const int e = threadIdx.x;
    const int b = blockIdx.x;

    float v[KMAX];
    int   c[KMAX];
#pragma unroll
    for (int k = 0; k < KMAX; ++k) {
        v[k] = vals[e * KMAX + k];
        c[k] = cols[e * KMAX + k];
    }
    hs[0][e] = 0.0f;   // h_0 = 0
    __syncthreads();

    const float* xrow = xin + (size_t)b * SEQ * EMB + e;
    int p = 0;
    for (int t = 0; t < SEQ; ++t) {
        float s = xrow[(size_t)t * EMB];
        const float* hp = hs[p];
#pragma unroll
        for (int k = 0; k < KMAX; ++k) s += v[k] * hp[c[k]];   // LDS gathers
        float h = tanhf(s);
        hs[p ^ 1][e] = h;          // write the *other* buffer: no read hazard
        __syncthreads();           // single barrier per step
        p ^= 1;
        size_t oi = ((size_t)b * SEQ + t) * EMB + e;
        if (write_bf) hout_bf[oi] = (bf16_t)h;
        else          hout_f32[oi] = h;
    }
}

// ---------------------------------------------------------------------------
// Kernel 5: readout GEMM with bf16 WMMA, f32 accumulate.
//   out[m][n] = sum_k A[m][k] * W[n][k] + bias[n]
// Workgroup: 256 threads = 8 waves. The 64x1024 bf16 A tile (128 KB,
// contiguous in global) is DMA'd into LDS by the Tensor Data Mover (one
// tensor_load_to_lds issued by wave 0; TENSORcnt waited, then barrier).
// Each wave: one 16-col N tile, FOUR 16-row M tiles -> each B fragment
// (2x global_load_b128) feeds 4 v_wmma_f32_16x16x32_bf16.
// Grid: x = NTOK/128 = 250 N-groups, y = 4096/64 = 64 M-groups.
//
// Fragment <-> VGPR layouts used here (16-bit, wave32; contiguous K, same order for A and B):
//   A (16x32): lane m = lane&15, g = lane>>4; half h: K = kk + 16g + h
//   B (32x16): lane n = lane&15; half h: K = kk + 16g + h
//   C/D (16x16 f32): vgpr r: M = r + 8g, N = lane&15.
// ---------------------------------------------------------------------------
__global__ __launch_bounds__(256) void gemm_wmma_kernel(
        const bf16_t* __restrict__ A,     // [4096][1024]
        const bf16_t* __restrict__ W,     // [32000][1024]
        const float*  __restrict__ bias,  // [32000]
        float*        __restrict__ out) { // [4096][32000]
    __shared__ __align__(16) bf16_t tileA[MROWS * EMB];   // 128 KB

    const int ng = blockIdx.x;    // N-group (128 cols)
    const int mg = blockIdx.y;    // M-group (64 rows)

    // ---- TDM: DMA the contiguous 128 KB A tile into LDS -------------------
    // 1-D tensor, 4-byte elements: 32768 elements = 128 KB.
    if (threadIdx.x < 32) {       // one wave issues the DMA (TDM ignores EXEC)
        const unsigned long long gaddr =
            (unsigned long long)(const void*)(A + (size_t)mg * MROWS * EMB);
        const unsigned lds_off = (unsigned)(size_t)(void*)tileA; // LDS byte offset
        const unsigned nelem   = MROWS * EMB / 2;                // 32768 dwords

        u32x4 g0;
        g0[0] = 1u;                                  // count=1, user descriptor
        g0[1] = lds_off;                             // lds_addr
        g0[2] = (unsigned)(gaddr & 0xFFFFFFFFu);     // global_addr[31:0]
        g0[3] = (unsigned)((gaddr >> 32) & 0x01FFFFFFu) | 0x80000000u; // [56:32] | type=2

        i32x8 g1;
        g1[0] = (int)(2u << 16);          // workgroup_mask=0, data_size=2 (4B)
        g1[1] = (int)((nelem & 0xFFFFu) << 16);          // tensor_dim0[15:0]
        g1[2] = (int)((nelem >> 16) | (1u << 16));       // tensor_dim0[31:16] | tensor_dim1=1
        g1[3] = (int)((nelem & 0xFFFFu) << 16);          // tile_dim0 = nelem (fits 16 bits)
        g1[4] = 0;                                        // tile_dim1/2 = 0 (unused)
        g1[5] = (int)nelem;                               // tensor_dim0_stride[31:0]
        g1[6] = 0;                                        // stride hi | dim1_stride lo
        g1[7] = 0;
        i32x4 gz4 = {0, 0, 0, 0};                         // groups 2/3: zero (<=2D)
        i32x8 gz8 = {0, 0, 0, 0, 0, 0, 0, 0};            // trailing group: zero

        // 6-arg clang-23 form: (g0, g1, g2, g3, g4, cpol)
        __builtin_amdgcn_tensor_load_to_lds(g0, g1, gz4, gz4, gz8, 0);
        __builtin_amdgcn_s_wait_tensorcnt(0);
    }
    __syncthreads();

    const int wave = threadIdx.x >> 5;
    const int lane = threadIdx.x & 31;
    const int g    = lane >> 4;        // lane half-group
    const int nl   = lane & 15;
    const int n    = ng * 128 + wave * 16 + nl;   // this lane's output column
    const int r    = nl;                          // A row within each M tile

    v8f acc0 = {}, acc1 = {}, acc2 = {}, acc3 = {};

    const uint4* bptr = (const uint4*)(W + (size_t)n * EMB);
    const uint4* l0   = (const uint4*)(tileA + (size_t)(r)      * EMB);
    const uint4* l1   = (const uint4*)(tileA + (size_t)(16 + r) * EMB);
    const uint4* l2   = (const uint4*)(tileA + (size_t)(32 + r) * EMB);
    const uint4* l3   = (const uint4*)(tileA + (size_t)(48 + r) * EMB);

    for (int kk = 0; kk < EMB; kk += 32) {
        // B fragment: halves = W[n][kk+16g .. kk+16g+15]
        FragBF bf;
        const int bq = (kk >> 3) + 2 * g;         // uint4 index (8 halves each)
        bf.q[0] = bptr[bq];
        bf.q[1] = bptr[bq + 1];

        // A fragments: halves 0..15 at K=kk+16g .. kk+16g+15
        FragBF a0, a1, a2, a3;
        const int aq = (kk + 16 * g) >> 3;
        a0.q[0] = l0[aq]; a0.q[1] = l0[aq + 1];
        a1.q[0] = l1[aq]; a1.q[1] = l1[aq + 1];
        a2.q[0] = l2[aq]; a2.q[1] = l2[aq + 1];
        a3.q[0] = l3[aq]; a3.q[1] = l3[aq + 1];

        // Pull next B chunk toward the caches (global_prefetch_b8).
        __builtin_prefetch((const void*)(W + (size_t)n * EMB + kk + 32), 0, 1);

        acc0 = __builtin_amdgcn_wmma_f32_16x16x32_bf16(
                   false, a0.v, false, bf.v, (short)0, acc0, false, false);
        acc1 = __builtin_amdgcn_wmma_f32_16x16x32_bf16(
                   false, a1.v, false, bf.v, (short)0, acc1, false, false);
        acc2 = __builtin_amdgcn_wmma_f32_16x16x32_bf16(
                   false, a2.v, false, bf.v, (short)0, acc2, false, false);
        acc3 = __builtin_amdgcn_wmma_f32_16x16x32_bf16(
                   false, a3.v, false, bf.v, (short)0, acc3, false, false);
    }

    const float bv = bias[n];
    float* orow = out + (size_t)(mg * MROWS) * NTOK + n;
#pragma unroll
    for (int rr = 0; rr < 8; ++rr) {
        const size_t row = (size_t)(rr + 8 * g);
        orow[(row)      * NTOK] = acc0[rr] + bv;   // M tile 0
        orow[(row + 16) * NTOK] = acc1[rr] + bv;   // M tile 1
        orow[(row + 32) * NTOK] = acc2[rr] + bv;   // M tile 2
        orow[(row + 48) * NTOK] = acc3[rr] + bv;   // M tile 3
    }
}

// ---------------------------------------------------------------------------
// Host launcher. Inputs (setup_inputs order):
//   d_in[0] x int32[4,1024]; d_in[1] emb f32[32000,1024];
//   d_in[2] a f32[2,1024,1024]; d_in[3] w_out f32[32000,1024];
//   d_in[4] b_out f32[32000].  Output: f32 [4,1024,32000].
// ---------------------------------------------------------------------------
extern "C" void kernel_launch(void* const* d_in, const int* in_sizes, int n_in,
                              void* d_out, int out_size, void* d_ws, size_t ws_size,
                              hipStream_t stream) {
    const int*   x     = (const int*)d_in[0];
    const float* emb   = (const float*)d_in[1];
    const float* a     = (const float*)d_in[2];
    const float* w_out = (const float*)d_in[3];
    const float* b_out = (const float*)d_in[4];
    float*       out   = (float*)d_out;

    // Workspace layout (all sizes 256B-multiples): ~108 MB total.
    char* ws = (char*)d_ws;
    float*  xe    = (float*)ws;  ws += (size_t)BATCH * SEQ * EMB * sizeof(float);   // 16 MB
    float*  h1    = (float*)ws;  ws += (size_t)BATCH * SEQ * EMB * sizeof(float);   // 16 MB
    bf16_t* h2bf  = (bf16_t*)ws; ws += (size_t)BATCH * SEQ * EMB * sizeof(bf16_t);  // 8 MB
    bf16_t* wbf   = (bf16_t*)ws; ws += (size_t)NTOK * EMB * sizeof(bf16_t);         // 65.5 MB
    float*  svals = (float*)ws;  ws += (size_t)LAYERS * EMB * KMAX * sizeof(float); // 128 KB
    int*    scols = (int*)ws;    ws += (size_t)LAYERS * EMB * KMAX * sizeof(int);   // 128 KB

    // 1) Sparse extraction of reservoir matrices.
    sparsify_kernel<<<(LAYERS * EMB) / 256, 256, 0, stream>>>(a, svals, scols);

    // 2) Embedding gather (independent of 1).
    gather_emb_kernel<<<(BATCH * SEQ * (EMB / 4)) / 256, 256, 0, stream>>>(x, emb, xe);

    // 3) w_out -> bf16 (independent of 1,2).
    cvt_bf16_kernel<<<(NTOK * EMB / 4) / 256, 256, 0, stream>>>(w_out, wbf);

    // 4) Two sequential reservoir layers (4 independent batch workgroups each).
    reservoir_layer_kernel<<<BATCH, EMB, 0, stream>>>(
        xe, svals, scols, h1, (bf16_t*)nullptr, 0);
    reservoir_layer_kernel<<<BATCH, EMB, 0, stream>>>(
        h1, svals + (size_t)EMB * KMAX, scols + (size_t)EMB * KMAX,
        (float*)nullptr, h2bf, 1);

    // 5) Readout GEMM with bf16 WMMA + TDM tile staging.
    dim3 grid(NTOK / 128, (BATCH * SEQ) / MROWS);
    gemm_wmma_kernel<<<grid, 256, 0, stream>>>(h2bf, wbf, b_out, out);
}
